// Head_64244120814309
// MI455X (gfx1250) — matrix operand order repaired and checked by reading, not compiled, not run on hardware
//
#include <hip/hip_runtime.h>
#include <math.h>

typedef _Float16 f16;
typedef __attribute__((ext_vector_type(16))) _Float16 v16h;
typedef __attribute__((ext_vector_type(2)))  _Float16 v2h;
typedef __attribute__((ext_vector_type(8)))  float    v8f;

#define N_EMBD 1024
#define HEAD   128
#define TSEQ   2048
#define NBATCH 16
#define MROWS  (NBATCH * TSEQ)   // 32768

__device__ __forceinline__ float hmax16(float x) {
  x = fmaxf(x, __shfl_xor(x, 8, 16));
  x = fmaxf(x, __shfl_xor(x, 4, 16));
  x = fmaxf(x, __shfl_xor(x, 2, 16));
  x = fmaxf(x, __shfl_xor(x, 1, 16));
  return x;
}
__device__ __forceinline__ float hsum16(float x) {
  x += __shfl_xor(x, 8, 16);
  x += __shfl_xor(x, 4, 16);
  x += __shfl_xor(x, 2, 16);
  x += __shfl_xor(x, 1, 16);
  return x;
}

// ---------------------------------------------------------------------------
// Kernel 0: one-time weight transpose+convert: WT[h][e] = (f16)W[e][h].
// ---------------------------------------------------------------------------
__global__ __launch_bounds__(256) void transpose_w(const float* __restrict__ Wk,
                                                   const float* __restrict__ Wq,
                                                   const float* __restrict__ Wv,
                                                   f16* __restrict__ WTk,
                                                   f16* __restrict__ WTq,
                                                   f16* __restrict__ WTv) {
  const float* W = (blockIdx.y == 0) ? Wk : (blockIdx.y == 1) ? Wq : Wv;
  f16*        WT = (blockIdx.y == 0) ? WTk : (blockIdx.y == 1) ? WTq : WTv;

  const int h  = threadIdx.x & 127;
  const int eo = threadIdx.x >> 7;
  const int e0 = blockIdx.x * 64 + eo * 32;
#pragma unroll
  for (int i = 0; i < 32; ++i) {
    const int e = e0 + i;
    WT[(long)h * N_EMBD + e] = (f16)W[(long)e * HEAD + h];  // reads coalesced
  }
}

// ---------------------------------------------------------------------------
// Kernel 1: out = (x @ W) * scale, f16 output, using pre-transposed f16 WT.
// 8 waves/block, each wave owns a 32-row M-tile (two A fragment groups) x
// all 128 cols, K unrolled by 2: per N-tile one 4x b128 clause (two 32B runs, 32B gap)
// fetches both K-chunk B-fragments, followed by 4 WMMAs (2 M x 2 K).
// transpose_out==1 writes the result transposed (vT[h][t]) via an LDS tile.
// ---------------------------------------------------------------------------
__global__ __launch_bounds__(256, 1) void proj_gemm(const float* __restrict__ x,
                                                    const f16* __restrict__ WT,
                                                    f16* __restrict__ out,
                                                    float scale, int transpose_out) {
  __shared__ f16 ldsT[HEAD * 256];  // output transpose staging [h][t], 64 KB

  const int tid  = threadIdx.x;
  const int wv   = tid >> 5;
  const int lane = tid & 31;
  const int hlf  = lane >> 4;
  const int l16  = lane & 15;
  const long m0b = (long)blockIdx.x * 256;
  const long m0  = m0b + wv * 32;

  const float* xrow0 = x + (m0 + l16) * (long)N_EMBD;
  const float* xrow1 = x + (m0 + 16 + l16) * (long)N_EMBD;
  const f16*  wtrow  = WT + (long)l16 * N_EMBD + hlf * 16;

  const v8f vzero = {0.f, 0.f, 0.f, 0.f, 0.f, 0.f, 0.f, 0.f};
  v8f acc[2][8];
#pragma unroll
  for (int g = 0; g < 2; ++g)
#pragma unroll
    for (int n = 0; n < 8; ++n) acc[g][n] = vzero;

  for (int kc = 0; kc < N_EMBD / 64; ++kc) {  // 64 K-elements per iteration
    if (kc + 2 < N_EMBD / 64) {
      __builtin_prefetch(xrow0 + (kc + 2) * 64, 0, 1);
      __builtin_prefetch(xrow1 + (kc + 2) * 64, 0, 1);
    }

    // Four A fragments: 2 rows x 2 K-chunks, from contiguous 64-float runs
    v16h a0e, a0o, a1e, a1o;
    {
      const float* px0 = xrow0 + kc * 64 + hlf * 16;
      const float* px1 = xrow1 + kc * 64 + hlf * 16;
#pragma unroll
      for (int i = 0; i < 8; ++i) {
        a0e[i] = (f16)px0[i];      a0e[8 + i] = (f16)px0[8 + i];
        a0o[i] = (f16)px0[32 + i]; a0o[8 + i] = (f16)px0[40 + i];
        a1e[i] = (f16)px1[i];      a1e[8 + i] = (f16)px1[8 + i];
        a1o[i] = (f16)px1[32 + i]; a1o[8 + i] = (f16)px1[40 + i];
      }
    }

#pragma unroll
    for (int n = 0; n < 8; ++n) {
      // Both K-chunk B-fragments for this N-tile: two 32B runs with a 32B gap -> one clause
      const f16* bp = wtrow + (long)n * 16 * N_EMBD + kc * 64;
      v16h bE = *(const v16h*)bp;
      v16h bO = *(const v16h*)(bp + 32);
      // 4 WMMAs per wait: two independent accumulate chains
      acc[0][n] = __builtin_amdgcn_wmma_f32_16x16x32_f16(
          false, a0e, false, bE, (short)0, acc[0][n], false, false);
      acc[1][n] = __builtin_amdgcn_wmma_f32_16x16x32_f16(
          false, a1e, false, bE, (short)0, acc[1][n], false, false);
      acc[0][n] = __builtin_amdgcn_wmma_f32_16x16x32_f16(
          false, a0o, false, bO, (short)0, acc[0][n], false, false);
      acc[1][n] = __builtin_amdgcn_wmma_f32_16x16x32_f16(
          false, a1o, false, bO, (short)0, acc[1][n], false, false);
    }
  }

  if (!transpose_out) {
    // C/D layout: component r -> row m0 + g*16 + r + 8*hlf, col = n*16 + l16
#pragma unroll
    for (int g = 0; g < 2; ++g)
#pragma unroll
      for (int n = 0; n < 8; ++n)
#pragma unroll
        for (int r = 0; r < 8; ++r)
          out[(m0 + g * 16 + r + hlf * 8) * HEAD + n * 16 + l16] =
              (f16)(acc[g][n][r] * scale);
  } else {
#pragma unroll
    for (int g = 0; g < 2; ++g)
#pragma unroll
      for (int n = 0; n < 8; ++n)
#pragma unroll
        for (int r = 0; r < 8; ++r)
          ldsT[(n * 16 + l16) * 256 + wv * 32 + g * 16 + r + hlf * 8] =
              (f16)(acc[g][n][r] * scale);
    __syncthreads();

    // coalesced transposed write-out: vT[(batch*HEAD + h)*TSEQ + t]
    const int h  = tid >> 1;
    const int sg = (tid & 1) * 128;
    const int batch = (int)(m0b / TSEQ);
    const long t0   = m0b - (long)batch * TSEQ;
    f16* dst = out + ((long)batch * HEAD + h) * TSEQ + t0 + sg;
    const f16* src = ldsT + h * 256 + sg;
#pragma unroll
    for (int i = 0; i < 16; ++i)
      *(uint4*)(dst + i * 8) = *(const uint4*)(src + i * 8);  // 8 halfs / 16B
  }
}

// ---------------------------------------------------------------------------
// Kernel 2: flash attention, one wave per 16-query tile, 32 keys/iteration.
// Keys even/odd interleaved so each lane's two probabilities are adjacent in
// the P-staging tile (one packed ds_store_b32). V-fragment loads issued
// before the softmax so their latency hides under the VALU work.
// ---------------------------------------------------------------------------
__global__ __launch_bounds__(256) void attn_flash(const f16* __restrict__ q,
                                                  const f16* __restrict__ k,
                                                  const f16* __restrict__ vT,
                                                  float* __restrict__ out) {
  __shared__ f16 pstage[8 * 512];  // 8 waves x (16x32 f16) = 8 KB

  const int wv   = threadIdx.x >> 5;
  const int lane = threadIdx.x & 31;
  const int hlf  = lane >> 4;
  const int l16  = lane & 15;

  const int gw    = blockIdx.x * 8 + wv;  // 0..2047 query tiles
  const int batch = gw >> 7;
  const int tile  = gw & 127;
  const long bbase = (long)batch * TSEQ;
  const int q0 = tile * 16;

  // Preload 4 Q A-fragments
  const f16* qrow = q + (bbase + q0 + l16) * (long)HEAD;
  v16h aq[4];
#pragma unroll
  for (int c = 0; c < 4; ++c) {
    const f16* p1 = qrow + c * 32 + hlf * 16;
    const f16* p2 = p1 + 8;
#pragma unroll
    for (int i = 0; i < 8; ++i) { aq[c][i] = p1[i]; aq[c][8 + i] = p2[i]; }
  }

  const v8f vzero = {0.f, 0.f, 0.f, 0.f, 0.f, 0.f, 0.f, 0.f};
  v8f acc[8];
#pragma unroll
  for (int n = 0; n < 8; ++n) acc[n] = vzero;

  float mrow[8], lrow[8];
#pragma unroll
  for (int r = 0; r < 8; ++r) { mrow[r] = -__builtin_inff(); lrow[r] = 0.f; }

  f16* ps = pstage + wv * 512;
  const f16* vbase = vT + (long)batch * HEAD * TSEQ;
  const int kmax = q0 + 16;

  for (int j = 0; j < kmax; j += 32) {
    // prefetch next iteration's K rows (one row per lane, 256B apart)
    if (j + 32 < kmax)
      __builtin_prefetch(k + (bbase + j + 32 + lane) * (long)HEAD, 0, 1);

    // ---- K^T fragments: S0 gets even keys j+2n, S1 gets odd keys j+2n+1 ----
    v16h kb[8];
#pragma unroll
    for (int c = 0; c < 4; ++c) {
      const f16* kr = k + (bbase + j + 2 * l16) * (long)HEAD + c * 32 + hlf * 16;
      kb[2 * c]     = *(const v16h*)kr;           // even key for this lane-col
      kb[2 * c + 1] = *(const v16h*)(kr + HEAD);  // odd key
    }
    // ---- V fragments issued early: latency hides under softmax VALU ----
    v16h bv[8];
#pragma unroll
    for (int n = 0; n < 8; ++n)
      bv[n] = *(const v16h*)(vbase + (long)(n * 16 + l16) * TSEQ + j + hlf * 16);

    // ---- S = Q . K^T ----
    v8f s0 = vzero, s1 = vzero;
#pragma unroll
    for (int c = 0; c < 4; ++c) {
      s0 = __builtin_amdgcn_wmma_f32_16x16x32_f16(false, aq[c], false, kb[2 * c],
                                                  (short)0, s0, false, false);
      s1 = __builtin_amdgcn_wmma_f32_16x16x32_f16(false, aq[c], false, kb[2 * c + 1],
                                                  (short)0, s1, false, false);
    }

    // ---- causal mask + online softmax (row = r + 8*hlf) ----
#pragma unroll
    for (int r = 0; r < 8; ++r) {
      const int qi = q0 + r + hlf * 8;
      const int ki = j + 2 * l16;
      float a0 = (ki     <= qi) ? s0[r] : -__builtin_inff();
      float a1 = (ki + 1 <= qi) ? s1[r] : -__builtin_inff();
      float mn    = fmaxf(mrow[r], hmax16(fmaxf(a0, a1)));
      float alpha = __expf(mrow[r] - mn);
      float p0    = __expf(a0 - mn);
      float p1    = __expf(a1 - mn);
      lrow[r] = lrow[r] * alpha + hsum16(p0 + p1);
      mrow[r] = mn;
#pragma unroll
      for (int n = 0; n < 8; ++n) acc[n][r] *= alpha;
      // P columns 2*l16 (even key) and 2*l16+1 (odd key): one packed store
      v2h pp; pp[0] = (f16)p0; pp[1] = (f16)p1;
      *(v2h*)(ps + (r + hlf * 8) * 32 + 2 * l16) = pp;
    }

    // ---- reload P as 16x32 A fragment (same-wave LDS is in-order) ----
    v16h ap;
    {
      const f16* p1 = ps + l16 * 32 + hlf * 16;
      const f16* p2 = p1 + 8;
#pragma unroll
      for (int i = 0; i < 8; ++i) { ap[i] = p1[i]; ap[8 + i] = p2[i]; }
    }

    // ---- O += P . V ----
#pragma unroll
    for (int n = 0; n < 8; ++n)
      acc[n] = __builtin_amdgcn_wmma_f32_16x16x32_f16(
          false, ap, false, bv[n], (short)0, acc[n], false, false);
  }

  // ---- normalize and write fp32 output ----
#pragma unroll
  for (int n = 0; n < 8; ++n)
#pragma unroll
    for (int r = 0; r < 8; ++r)
      out[(bbase + q0 + r + hlf * 8) * HEAD + n * 16 + l16] =
          acc[n][r] / lrow[r];
}

// ---------------------------------------------------------------------------
extern "C" void kernel_launch(void* const* d_in, const int* in_sizes, int n_in,
                              void* d_out, int out_size, void* d_ws, size_t ws_size,
                              hipStream_t stream) {
  (void)in_sizes; (void)n_in; (void)out_size; (void)ws_size;
  const float* x  = (const float*)d_in[0];
  const float* Wk = (const float*)d_in[1];
  const float* Wq = (const float*)d_in[2];
  const float* Wv = (const float*)d_in[3];
  float* out = (float*)d_out;

  // workspace layout (f16): k | q (pre-scaled) | vT | WTk | WTq | WTv
  f16* kbuf = (f16*)d_ws;
  f16* qbuf = kbuf + (size_t)MROWS * HEAD;
  f16* vbuf = qbuf + (size_t)MROWS * HEAD;   // holds vT[batch][h][t]
  f16* wtk  = vbuf + (size_t)MROWS * HEAD;
  f16* wtq  = wtk + (size_t)HEAD * N_EMBD;
  f16* wtv  = wtq + (size_t)HEAD * N_EMBD;

  const float qscale = 0.08838834764831845f;  // 1/sqrt(128)

  dim3 blk(256);
  transpose_w<<<dim3(16, 3), blk, 0, stream>>>(Wk, Wq, Wv, wtk, wtq, wtv);

  dim3 grdP(MROWS / 256);  // 128 blocks, 8 waves x 32 rows each
  proj_gemm<<<grdP, blk, 0, stream>>>(x, wtk, kbuf, 1.0f, 0);
  proj_gemm<<<grdP, blk, 0, stream>>>(x, wtq, qbuf, qscale, 0);
  proj_gemm<<<grdP, blk, 0, stream>>>(x, wtv, vbuf, 1.0f, 1);

  dim3 grdA((NBATCH * TSEQ / 16) / 8);  // 256 blocks
  attn_flash<<<grdA, blk, 0, stream>>>(qbuf, kbuf, vbuf, out);
}
